// LazarusBlock_24670292148234
// MI455X (gfx1250) — compile-verified, matrix operand order checked by reading, not run on hardware
//
#include <hip/hip_runtime.h>
#include <hip/hip_bf16.h>

// ---------------------------------------------------------------------------
// Lazarus block for MI455X (gfx1250): bf16 WMMA GEMMs + async-LDS staging +
// VALU selective-scan pipeline.
// ---------------------------------------------------------------------------

#define B_ 4
#define C_ 2048
#define D_ 1024
#define DCONV_ 512
#define DMAMBA_ 512
#define KCONV_ 4
#define DSTATE_ 16
#define FFN_ 2560
#define ROWS_ (B_ * C_)   // 8192 tokens

typedef __bf16 v16bf __attribute__((ext_vector_type(16)));
typedef float v8f __attribute__((ext_vector_type(8)));
typedef unsigned short u16x8 __attribute__((ext_vector_type(8)));

// gfx1250 async global->LDS copy (ASYNCcnt-tracked, bypasses VGPRs).
#if defined(__has_builtin)
#  if __has_builtin(__builtin_amdgcn_global_load_async_to_lds_b128) && \
      __has_builtin(__builtin_amdgcn_s_wait_asynccnt)
#    define HAVE_ASYNC_LDS 1
#  endif
#endif
#ifndef HAVE_ASYNC_LDS
#  define HAVE_ASYNC_LDS 0
#endif

#if HAVE_ASYNC_LDS
// The builtin expects pointers to a 16-byte int vector (GCC vector_size style),
// global (addrspace 1) source and LDS (addrspace 3) destination.
typedef int v4i_gcc __attribute__((vector_size(16)));
typedef __attribute__((address_space(1))) v4i_gcc* gvp_t;
typedef __attribute__((address_space(3))) v4i_gcc* lvp_t;
#endif

__device__ __forceinline__ unsigned short f2bf(float f) {
    unsigned u = __builtin_bit_cast(unsigned, f);
    unsigned r = (u + 0x7FFFu + ((u >> 16) & 1u)) >> 16;   // round-to-nearest-even
    return (unsigned short)r;
}
__device__ __forceinline__ float sigmoidf_(float x) { return 1.f / (1.f + expf(-x)); }
__device__ __forceinline__ float siluf_(float x) { return x * sigmoidf_(x); }
__device__ __forceinline__ float softplusf_(float x) { return (x > 20.f) ? x : log1pf(expf(x)); }

// ---------------------------------------------------------------------------
// Generic strided GEMM: C[M,N](f32 or bf16) = A[M,K](bf16) * B[K,N](bf16)
// Block: 256 threads (8 waves). Tile 128(M) x 64(N), BK=32 (one WMMA K-step).
// Wave grid 4x2; each wave computes a 32x32 patch = 2x2 WMMA 16x16 tiles.
// FAST=true assumes sAk==1 and sBn==1 (contiguous K rows / N cols):
//   A tile is staged with GLOBAL_LOAD_ASYNC_TO_LDS_B128 (or vector loads),
//   B tile with 128-bit loads + in-LDS transpose.
// Requires: M % 128 == 0, N % 64 == 0, K % 32 == 0 (true for all call sites).
// ---------------------------------------------------------------------------
template <bool WRITE_BF16, bool FAST>
__global__ __launch_bounds__(256)
void gemm_bf16_wmma(const unsigned short* __restrict__ A, long sAm, long sAk,
                    const unsigned short* __restrict__ B, long sBk, long sBn,
                    void* __restrict__ Cout, int M, int N, int K)
{
    constexpr int BM = 128, BN = 64, BK = 32;
    __shared__ __align__(16) unsigned short As[BM * BK];    // [m][k]
    __shared__ __align__(16) unsigned short Bst[BN * BK];   // transposed: [n][k]

    const int tid   = threadIdx.x;
    const int wid   = tid >> 5;          // wave32
    const int lane  = tid & 31;
    const int lm    = lane & 15;         // row (A) / col (B,D) within 16x16
    const int lh    = lane >> 4;         // half select
    const int waveM = wid >> 1;          // 0..3
    const int waveN = wid & 1;           // 0..1
    const int m0    = blockIdx.y * BM;
    const int n0    = blockIdx.x * BN;

    v8f acc[2][2];
#pragma unroll
    for (int i = 0; i < 2; ++i)
#pragma unroll
        for (int j = 0; j < 2; ++j)
#pragma unroll
            for (int r = 0; r < 8; ++r) acc[i][j][r] = 0.0f;

    union FragU { u16x8 h[2]; v16bf v; };

    for (int k0 = 0; k0 < K; k0 += BK) {
        if (FAST) {
            // ---- A tile: 512 chunks of 16B, contiguous global -> contiguous LDS
#pragma unroll
            for (int q = tid; q < (BM * BK) / 8; q += 256) {
                int r  = q >> 2;          // row 0..127
                int c8 = (q & 3) << 3;    // k sub-offset 0/8/16/24
                const unsigned short* src = A + (long)(m0 + r) * sAm + (k0 + c8);
#if HAVE_ASYNC_LDS
                __builtin_amdgcn_global_load_async_to_lds_b128(
                    (gvp_t)(uintptr_t)(const void*)src,
                    (lvp_t)(unsigned)(uintptr_t)&As[r * BK + c8], 0, 0);
#else
                *(u16x8*)&As[r * BK + c8] = *(const u16x8*)src;
#endif
                if (k0 + BK < K) __builtin_prefetch(src + BK, 0, 1);
            }
            // ---- B tile: 256 chunks of 16B, transpose into LDS ([n][k])
            {
                int r  = tid >> 3;        // k row 0..31
                int c8 = (tid & 7) << 3;  // n sub-offset
                const unsigned short* src = B + (long)(k0 + r) * sBk + (n0 + c8);
                u16x8 v = *(const u16x8*)src;
#pragma unroll
                for (int j = 0; j < 8; ++j) Bst[(c8 + j) * BK + r] = v[j];
                if (k0 + BK < K) __builtin_prefetch(src + BK * sBk, 0, 1);
            }
#if HAVE_ASYNC_LDS
            __builtin_amdgcn_s_wait_asynccnt(0);
#endif
        } else {
            // ---- generic strided staging (used for the transposed-A einsum)
            for (int i = tid; i < BM * BK; i += 256) {
                int r = i >> 5, c = i & 31;
                As[i] = A[(long)(m0 + r) * sAm + (long)(k0 + c) * sAk];
            }
            for (int i = tid; i < BK * BN; i += 256) {
                int r = i >> 6, c = i & 63;   // r = k, c = n
                Bst[c * BK + r] = B[(long)(k0 + r) * sBk + (long)(n0 + c) * sBn];
            }
        }
        __syncthreads();

        // Fragment layout (ISA 7.12.2, 16-bit A 16x32): lane element j maps to
        // k = 16*(j>>3) + 8*lh + (j&7)  -> two contiguous 8-element runs.
        v16bf afrag[2], bfrag[2];
#pragma unroll
        for (int i = 0; i < 2; ++i) {
            int am = waveM * 32 + i * 16 + lm;
            FragU f;
            f.h[0] = *(const u16x8*)&As[am * BK + lh * 8];
            f.h[1] = *(const u16x8*)&As[am * BK + 16 + lh * 8];
            afrag[i] = f.v;
        }
#pragma unroll
        for (int j = 0; j < 2; ++j) {
            int bn = waveN * 32 + j * 16 + lm;
            FragU f;
            f.h[0] = *(const u16x8*)&Bst[bn * BK + lh * 8];
            f.h[1] = *(const u16x8*)&Bst[bn * BK + 16 + lh * 8];
            bfrag[j] = f.v;
        }
#pragma unroll
        for (int i = 0; i < 2; ++i)
#pragma unroll
            for (int j = 0; j < 2; ++j)
                acc[i][j] = __builtin_amdgcn_wmma_f32_16x16x32_bf16(
                    false, afrag[i], false, bfrag[j], (short)0, acc[i][j], false, false);
        __syncthreads();
    }

    // D layout: VGPR r -> row r + 8*lh, col lm.
#pragma unroll
    for (int i = 0; i < 2; ++i)
#pragma unroll
        for (int j = 0; j < 2; ++j) {
            int mrow0 = m0 + waveM * 32 + i * 16 + lh * 8;
            int ncol  = n0 + waveN * 32 + j * 16 + lm;
#pragma unroll
            for (int r = 0; r < 8; ++r) {
                long idx = (long)(mrow0 + r) * N + ncol;
                float val = acc[i][j][r];
                if (WRITE_BF16) ((unsigned short*)Cout)[idx] = f2bf(val);
                else            ((float*)Cout)[idx] = val;
            }
        }
}

// ---------------------------------------------------------------------------
// RMS norm: per-row (D=1024), writes fp32 (optional) + bf16.
// ---------------------------------------------------------------------------
__global__ __launch_bounds__(256)
void rmsnorm_kernel(const float* __restrict__ x, const float* __restrict__ w,
                    float* __restrict__ out_f32, unsigned short* __restrict__ out_bf)
{
    __shared__ float red[256];
    long row = blockIdx.x;
    const float* xr = x + row * D_;
    float s = 0.f;
    for (int d = threadIdx.x; d < D_; d += 256) { float v = xr[d]; s += v * v; }
    red[threadIdx.x] = s; __syncthreads();
    for (int off = 128; off > 0; off >>= 1) {
        if (threadIdx.x < off) red[threadIdx.x] += red[threadIdx.x + off];
        __syncthreads();
    }
    float scale = rsqrtf(red[0] * (1.0f / D_) + 1e-6f);
    for (int d = threadIdx.x; d < D_; d += 256) {
        float v = xr[d] * scale * w[d];
        if (out_f32) out_f32[row * D_ + d] = v;
        out_bf[row * D_ + d] = f2bf(v);
    }
}

// f32 -> bf16 convert
__global__ void cvt_bf16_kernel(const float* __restrict__ src, unsigned short* __restrict__ dst, long n)
{
    long i = (long)blockIdx.x * blockDim.x + threadIdx.x;
    if (i < n) dst[i] = f2bf(src[i]);
}

// (a + b) -> bf16   (merged ffn_down_w + delta_W)
__global__ void addcvt_bf16_kernel(const float* __restrict__ a, const float* __restrict__ b,
                                   unsigned short* __restrict__ dst, long n)
{
    long i = (long)blockIdx.x * blockDim.x + threadIdx.x;
    if (i < n) dst[i] = f2bf(a[i] + b[i]);
}

// ---------------------------------------------------------------------------
// B/C small projections: BsCs[row][0..15]=xn@w_B, [16..31]=xn@w_C.
// One block per row; 8 K-chunks x 32 outputs, LDS reduce.
// ---------------------------------------------------------------------------
__global__ __launch_bounds__(256)
void bc_proj_kernel(const float* __restrict__ xn, const float* __restrict__ wB,
                    const float* __restrict__ wC, float* __restrict__ BsCs)
{
    __shared__ float part[256];
    long row = blockIdx.x;
    int t = threadIdx.x;
    int j = t & 31;
    int chunk = t >> 5;
    int n = j & 15;
    const float* xr = xn + row * D_;
    float s = 0.f;
    int k0 = chunk * 128;
    for (int k = k0; k < k0 + 128; ++k)
        s += xr[k] * ((j < 16) ? wB[k * 16 + n] : wC[k * 16 + n]);
    part[t] = s; __syncthreads();
    if (t < 32) {
        float acc = 0.f;
#pragma unroll
        for (int c = 0; c < 8; ++c) acc += part[c * 32 + t];
        BsCs[row * 32 + t] = acc;
    }
}

// Depthwise causal conv (K=4) + silu gate. gv: [row][0..511]=gate, [512..1023]=value.
__global__ void conv_silu_kernel(const float* __restrict__ gv, const float* __restrict__ kw,
                                 float* __restrict__ out)
{
    int i = blockIdx.x * blockDim.x + threadIdx.x;
    if (i >= ROWS_ * DCONV_) return;
    int d = i & (DCONV_ - 1);
    long row = (long)(i >> 9);
    int c = (int)(row & (C_ - 1));
    float acc = 0.f;
#pragma unroll
    for (int k = 0; k < KCONV_; ++k) {
        int cc = c - (KCONV_ - 1) + k;
        if (cc >= 0)
            acc += gv[(row - (KCONV_ - 1) + k) * (2 * DCONV_) + DCONV_ + d] * kw[k * DCONV_ + d];
    }
    float g = gv[row * (2 * DCONV_) + d];
    out[row * DCONV_ + d] = siluf_(g) * acc;
}

// ---------------------------------------------------------------------------
// Selective scan: one thread per (b,d) channel; 16 states in registers.
// Block covers a single batch b (256 channels); B/C row staged in LDS.
// ---------------------------------------------------------------------------
__global__ __launch_bounds__(256)
void ssm_scan_kernel(const float* __restrict__ uz, const float* __restrict__ dtr,
                     const float* __restrict__ dtb, const float* __restrict__ Alog,
                     const float* __restrict__ BsCs, const float* __restrict__ Dp,
                     float* __restrict__ out)
{
    __shared__ float bc[32];
    int t = threadIdx.x;
    int b = blockIdx.x >> 1;
    int d = ((blockIdx.x & 1) << 8) + t;
    float A[DSTATE_], h[DSTATE_];
#pragma unroll
    for (int s = 0; s < DSTATE_; ++s) { A[s] = -expf(Alog[d * DSTATE_ + s]); h[s] = 0.f; }
    float bias = dtb[d];
    float Dv = Dp[d];
    for (int c = 0; c < C_; ++c) {
        long row = (long)b * C_ + c;
        if (t < 32) bc[t] = BsCs[row * 32 + t];
        __syncthreads();
        float dtv = softplusf_(dtr[row * DMAMBA_ + d] + bias);
        float u = uz[row * (2 * DMAMBA_) + d];
        float zv = uz[row * (2 * DMAMBA_) + DMAMBA_ + d];
        float du = dtv * u;
        float y = 0.f;
#pragma unroll
        for (int s = 0; s < DSTATE_; ++s) {
            h[s] = expf(dtv * A[s]) * h[s] + du * bc[s];
            y += h[s] * bc[16 + s];
        }
        out[row * DMAMBA_ + d] = (y + Dv * u) * siluf_(zv);
        __syncthreads();
    }
}

// mixed = concat(conv_out, ssm_out) -> bf16
__global__ void mix_bf16_kernel(const float* __restrict__ conv, const float* __restrict__ ssm,
                                unsigned short* __restrict__ mixed)
{
    long i = (long)blockIdx.x * blockDim.x + threadIdx.x;
    if (i >= (long)ROWS_ * D_) return;
    long row = i >> 10; int c = (int)(i & 1023);
    float v = (c < DCONV_) ? conv[row * DCONV_ + c] : ssm[row * DMAMBA_ + (c - DCONV_)];
    mixed[i] = f2bf(v);
}

// velocity = sigmoid(log_beta)*velocity + mixed_out;  x_mid = x + velocity
__global__ void resid_kernel(const float* __restrict__ x, const float* __restrict__ vel,
                             const float* __restrict__ mo, const float* __restrict__ log_beta,
                             float* __restrict__ vel_out, float* __restrict__ x_mid, long n)
{
    long i = (long)blockIdx.x * blockDim.x + threadIdx.x;
    if (i >= n) return;
    float beta = sigmoidf_(log_beta[0]);
    float v = beta * vel[i] + mo[i];
    vel_out[i] = v;
    x_mid[i] = x[i] + v;
}

// z = silu(gate)*up  -> bf16
__global__ void swiglu_kernel(const float* __restrict__ gu, unsigned short* __restrict__ z, long n)
{
    long i = (long)blockIdx.x * blockDim.x + threadIdx.x;
    if (i >= n) return;
    long row = i / FFN_; int f = (int)(i % FFN_);
    float g = gu[row * (2 * FFN_) + f];
    float u = gu[row * (2 * FFN_) + FFN_ + f];
    z[i] = f2bf(siluf_(g) * u);
}

// x_out = x_mid + ffn_out
__global__ void finalx_kernel(const float* __restrict__ x_mid, const float* __restrict__ ffn_out,
                              float* __restrict__ x_out, long n)
{
    long i = (long)blockIdx.x * blockDim.x + threadIdx.x;
    if (i < n) x_out[i] = x_mid[i] + ffn_out[i];
}

// new_delta_W = sigmoid(log_gamma)*delta_W + softplus(log_eta)*update/B
__global__ void deltaw_kernel(const float* __restrict__ dW, const float* __restrict__ upd,
                              const float* __restrict__ log_gamma, const float* __restrict__ log_eta,
                              float* __restrict__ out, long n)
{
    long i = (long)blockIdx.x * blockDim.x + threadIdx.x;
    if (i >= n) return;
    float gamma = sigmoidf_(log_gamma[0]);
    float eta = softplusf_(log_eta[0]) * (1.0f / B_);
    out[i] = gamma * dW[i] + eta * upd[i];
}

// ---------------------------------------------------------------------------
// Host-side orchestration
// ---------------------------------------------------------------------------
extern "C" void kernel_launch(void* const* d_in, const int* in_sizes, int n_in,
                              void* d_out, int out_size, void* d_ws, size_t ws_size,
                              hipStream_t stream)
{
    (void)in_sizes; (void)n_in; (void)out_size; (void)ws_size;
    const float* x        = (const float*)d_in[0];
    const float* velocity = (const float*)d_in[1];
    const float* v_hat    = (const float*)d_in[2];
    const float* delta_W  = (const float*)d_in[3];
    const float* pre_w    = (const float*)d_in[4];
    const float* ffn_w    = (const float*)d_in[5];
    const float* conv_in  = (const float*)d_in[6];
    const float* conv_k   = (const float*)d_in[7];
    const float* ssm_in   = (const float*)d_in[8];
    const float* ssm_dt   = (const float*)d_in[9];
    const float* dt_b     = (const float*)d_in[10];
    const float* A_log    = (const float*)d_in[11];
    const float* w_B      = (const float*)d_in[12];
    const float* w_C      = (const float*)d_in[13];
    const float* ssm_D    = (const float*)d_in[14];
    const float* out_proj = (const float*)d_in[15];
    const float* log_beta = (const float*)d_in[16];
    const float* gate_up  = (const float*)d_in[17];
    const float* ffn_down = (const float*)d_in[18];
    const float* ffn_tgt  = (const float*)d_in[19];
    const float* log_gamma= (const float*)d_in[20];
    const float* log_eta  = (const float*)d_in[21];

    float* out_x   = (float*)d_out;                        // 8192*1024
    float* out_vel = out_x + (long)ROWS_ * D_;             // 8192*1024
    float* out_dW  = out_vel + (long)ROWS_ * D_;           // 2560*1024

    // Workspace carving
    char* base = (char*)d_ws;
    size_t cur = 0;
    auto alloc = [&](size_t bytes) -> char* {
        char* p = base + cur;
        cur = (cur + bytes + 255) & ~(size_t)255;
        return p;
    };
    float*          xn_f32    = (float*)alloc((size_t)ROWS_ * D_ * 4);
    unsigned short* xn_bf     = (unsigned short*)alloc((size_t)ROWS_ * D_ * 2);
    unsigned short* w_conv_bf = (unsigned short*)alloc((size_t)D_ * 2 * DCONV_ * 2);
    unsigned short* w_ssmin_bf= (unsigned short*)alloc((size_t)D_ * 2 * DMAMBA_ * 2);
    unsigned short* w_dt_bf   = (unsigned short*)alloc((size_t)D_ * DMAMBA_ * 2);
    unsigned short* w_outp_bf = (unsigned short*)alloc((size_t)D_ * D_ * 2);
    unsigned short* w_gu_bf   = (unsigned short*)alloc((size_t)D_ * 2 * FFN_ * 2);
    unsigned short* w_tgt_bf  = (unsigned short*)alloc((size_t)D_ * FFN_ * 2);
    unsigned short* w_comb_bf = (unsigned short*)alloc((size_t)FFN_ * D_ * 2);
    unsigned short* vhat_bf   = (unsigned short*)alloc((size_t)ROWS_ * D_ * 2);
    float*          gv_conv   = (float*)alloc((size_t)ROWS_ * 2 * DCONV_ * 4);
    float*          uz        = (float*)alloc((size_t)ROWS_ * 2 * DMAMBA_ * 4);
    float*          dt_raw    = (float*)alloc((size_t)ROWS_ * DMAMBA_ * 4);
    float*          BsCs      = (float*)alloc((size_t)ROWS_ * 32 * 4);
    float*          conv_out  = (float*)alloc((size_t)ROWS_ * DCONV_ * 4);
    float*          ssm_out   = (float*)alloc((size_t)ROWS_ * DMAMBA_ * 4);
    unsigned short* mixed_bf  = (unsigned short*)alloc((size_t)ROWS_ * D_ * 2);
    float*          mixed_out = (float*)alloc((size_t)ROWS_ * D_ * 4);
    float*          x_mid     = (float*)alloc((size_t)ROWS_ * D_ * 4);
    unsigned short* ffn_in_bf = (unsigned short*)alloc((size_t)ROWS_ * D_ * 2);
    float*          gu_out    = (float*)alloc((size_t)ROWS_ * 2 * FFN_ * 4);
    unsigned short* z_bf      = (unsigned short*)alloc((size_t)ROWS_ * FFN_ * 2);
    float*          ffn_out   = (float*)alloc((size_t)ROWS_ * D_ * 4);
    unsigned short* vproj_bf  = (unsigned short*)alloc((size_t)ROWS_ * FFN_ * 2);
    float*          update    = (float*)alloc((size_t)FFN_ * D_ * 4);

    auto blocks = [](long n) { return (unsigned)((n + 255) / 256); };

    // 1) pre-norm
    rmsnorm_kernel<<<ROWS_, 256, 0, stream>>>(x, pre_w, xn_f32, xn_bf);

    // 2) weight converts (bf16), merged down-proj weight, v_hat bf16
    cvt_bf16_kernel<<<blocks((long)D_ * 2 * DCONV_), 256, 0, stream>>>(conv_in, w_conv_bf, (long)D_ * 2 * DCONV_);
    cvt_bf16_kernel<<<blocks((long)D_ * 2 * DMAMBA_), 256, 0, stream>>>(ssm_in, w_ssmin_bf, (long)D_ * 2 * DMAMBA_);
    cvt_bf16_kernel<<<blocks((long)D_ * DMAMBA_), 256, 0, stream>>>(ssm_dt, w_dt_bf, (long)D_ * DMAMBA_);
    cvt_bf16_kernel<<<blocks((long)D_ * D_), 256, 0, stream>>>(out_proj, w_outp_bf, (long)D_ * D_);
    cvt_bf16_kernel<<<blocks((long)D_ * 2 * FFN_), 256, 0, stream>>>(gate_up, w_gu_bf, (long)D_ * 2 * FFN_);
    cvt_bf16_kernel<<<blocks((long)D_ * FFN_), 256, 0, stream>>>(ffn_tgt, w_tgt_bf, (long)D_ * FFN_);
    addcvt_bf16_kernel<<<blocks((long)FFN_ * D_), 256, 0, stream>>>(ffn_down, delta_W, w_comb_bf, (long)FFN_ * D_);
    cvt_bf16_kernel<<<blocks((long)ROWS_ * D_), 256, 0, stream>>>(v_hat, vhat_bf, (long)ROWS_ * D_);

    // 3) input projections (WMMA)
    gemm_bf16_wmma<false, true><<<dim3((2 * DCONV_) / 64, ROWS_ / 128), 256, 0, stream>>>(
        xn_bf, D_, 1, w_conv_bf, 2 * DCONV_, 1, gv_conv, ROWS_, 2 * DCONV_, D_);
    gemm_bf16_wmma<false, true><<<dim3((2 * DMAMBA_) / 64, ROWS_ / 128), 256, 0, stream>>>(
        xn_bf, D_, 1, w_ssmin_bf, 2 * DMAMBA_, 1, uz, ROWS_, 2 * DMAMBA_, D_);
    gemm_bf16_wmma<false, true><<<dim3(DMAMBA_ / 64, ROWS_ / 128), 256, 0, stream>>>(
        xn_bf, D_, 1, w_dt_bf, DMAMBA_, 1, dt_raw, ROWS_, DMAMBA_, D_);
    bc_proj_kernel<<<ROWS_, 256, 0, stream>>>(xn_f32, w_B, w_C, BsCs);

    // 4) gated conv + selective scan
    conv_silu_kernel<<<blocks((long)ROWS_ * DCONV_), 256, 0, stream>>>(gv_conv, conv_k, conv_out);
    ssm_scan_kernel<<<(B_ * DMAMBA_) / 256, 256, 0, stream>>>(uz, dt_raw, dt_b, A_log, BsCs, ssm_D, ssm_out);

    // 5) out-proj + momentum residual
    mix_bf16_kernel<<<blocks((long)ROWS_ * D_), 256, 0, stream>>>(conv_out, ssm_out, mixed_bf);
    gemm_bf16_wmma<false, true><<<dim3(D_ / 64, ROWS_ / 128), 256, 0, stream>>>(
        mixed_bf, D_, 1, w_outp_bf, D_, 1, mixed_out, ROWS_, D_, D_);
    resid_kernel<<<blocks((long)ROWS_ * D_), 256, 0, stream>>>(
        x, velocity, mixed_out, log_beta, out_vel, x_mid, (long)ROWS_ * D_);

    // 6) FFN norm + SwiGLU + merged down-proj
    rmsnorm_kernel<<<ROWS_, 256, 0, stream>>>(x_mid, ffn_w, nullptr, ffn_in_bf);
    gemm_bf16_wmma<false, true><<<dim3((2 * FFN_) / 64, ROWS_ / 128), 256, 0, stream>>>(
        ffn_in_bf, D_, 1, w_gu_bf, 2 * FFN_, 1, gu_out, ROWS_, 2 * FFN_, D_);
    swiglu_kernel<<<blocks((long)ROWS_ * FFN_), 256, 0, stream>>>(gu_out, z_bf, (long)ROWS_ * FFN_);
    gemm_bf16_wmma<false, true><<<dim3(D_ / 64, ROWS_ / 128), 256, 0, stream>>>(
        z_bf, FFN_, 1, w_comb_bf, D_, 1, ffn_out, ROWS_, D_, FFN_);
    finalx_kernel<<<blocks((long)ROWS_ * D_), 256, 0, stream>>>(x_mid, ffn_out, out_x, (long)ROWS_ * D_);

    // 7) fast-weight update: vproj = v_hat @ ffn_target (bf16 out),
    //    update[f,d] = sum_rows vproj[row,f] * ffn_in[row,d]  (A accessed transposed)
    gemm_bf16_wmma<true, true><<<dim3(FFN_ / 64, ROWS_ / 128), 256, 0, stream>>>(
        vhat_bf, D_, 1, w_tgt_bf, FFN_, 1, vproj_bf, ROWS_, FFN_, D_);
    gemm_bf16_wmma<false, false><<<dim3(D_ / 64, FFN_ / 128), 256, 0, stream>>>(
        vproj_bf, 1, FFN_, ffn_in_bf, D_, 1, update, FFN_, D_, ROWS_);
    deltaw_kernel<<<blocks((long)FFN_ * D_), 256, 0, stream>>>(
        delta_W, update, log_gamma, log_eta, out_dW, (long)FFN_ * D_);
}
